// FFD_77506979823873
// MI455X (gfx1250) — matrix operand order repaired and checked by reading, not compiled, not run on hardware
//
#include <hip/hip_runtime.h>

typedef __attribute__((ext_vector_type(16))) _Float16 v16h;
typedef __attribute__((ext_vector_type(2)))  _Float16 h2;
typedef __attribute__((ext_vector_type(8)))  float    v8f;

union BReg { v16h v; h2 p[8]; };
union DwH2 { unsigned int u; h2 h; };

// Degree-5 Bernstein basis: b[i] = C(5,i) s^i (1-s)^(5-i)
__device__ __forceinline__ void bern5(float s, float b[6]) {
  float u  = 1.0f - s;
  float s2 = s * s, s3 = s2 * s, s4 = s2 * s2, s5 = s4 * s;
  float u2 = u * u, u3 = u2 * u, u4 = u2 * u2, u5 = u4 * u;
  b[0] = u5;
  b[1] = 5.0f * s * u4;
  b[2] = 10.0f * s2 * u3;
  b[3] = 10.0f * s3 * u2;
  b[4] = 5.0f * s4 * u;
  b[5] = s5;
}

// D(16x16) = A(16x224 f16, rows = channel c) x B(224x16 f16, cols = vertices)
// K-slot permutation: K = 32t + 16h + e  ->  f = 16t + e,  ij = f % 36,
// k = f/36 + 3h. Adjacent (even) f pairs share the z-basis factor and have
// consecutive ij, so each packed B dword is one v_pk_mul_f16.
__global__ __launch_bounds__(256) void ffd_bernstein_wmma(
    const float* __restrict__ verts,   // [N,3]
    const float* __restrict__ ctrl,    // [3,6,6,6]
    float* __restrict__ out,           // [1,N,3]
    int N, int numTiles) {
  const int lane = threadIdx.x & 31;
  const int half = lane >> 4;   // which 16-wide K slice this lane supplies
  const int col  = lane & 15;   // B/D: vertex within tile; A rows = c

  // ---- Build the (wave-invariant) A register image once per block in LDS ---
  // Image: [chunk t][lane][8 dwords]  (7*32*8 dwords = 7 KiB)
  __shared__ alignas(16) unsigned int Aimg[7 * 32 * 8];
  for (int D = threadIdx.x; D < 7 * 32 * 8; D += 256) {
    const int g     = D & 7;          // dword within lane's chunk
    const int laneI = (D >> 3) & 31;  // target lane
    const int t     = D >> 8;         // chunk
    const int cI    = laneI & 15;     // A row (channel)
    const int hL    = laneI >> 4;
    DwH2 dw; dw.u = 0;
    if (cI < 3) {
      float v01[2] = {0.0f, 0.0f};
      for (int s = 0; s < 2; ++s) {
        // A layout matches B: lanes 0-15 hold K = kb+0..15,
        // lanes 16-31 hold K = kb+16..31 (element e -> f = 16t+e, z half = hL).
        const int e  = 2 * g + s;
        const int f  = 16 * t + e;
        const int hB = hL;
        if (f < 108) {
          const int ij = f % 36;            // = i*6 + j
          const int kk = f / 36 + 3 * hB;   // z index
          v01[s] = ctrl[cI * 216 + ij * 6 + kk];
        }
      }
      dw.h = h2{(_Float16)v01[0], (_Float16)v01[1]};
    }
    Aimg[D] = dw.u;
  }
  __syncthreads();

  // Each lane pulls its A registers: 2x ds_load_b128 per chunk.
  v16h A[7];
#pragma unroll
  for (int t = 0; t < 7; ++t)
    A[t] = *(const v16h*)(&Aimg[t * 256 + lane * 8]);

  const int wavesPerBlock = blockDim.x >> 5;
  const int waveId   = blockIdx.x * wavesPerBlock + (threadIdx.x >> 5);
  const int numWaves = gridDim.x * wavesPerBlock;

  for (int tile = waveId; tile < numTiles; tile += numWaves) {
    const int v  = tile * 16 + col;
    const int vc = (v < N) ? v : (N - 1);   // clamp: keep EXEC full for WMMA

    // Prefetch next tile's vertex data toward the caches.
    {
      const int nv = (tile + numWaves) * 16 + col;
      if (nv < N) __builtin_prefetch(verts + 3 * nv, 0, 0);
    }

    const float x = verts[3 * vc + 0];
    const float y = verts[3 * vc + 1];
    const float z = verts[3 * vc + 2];

    float bx[6], by[6], bz[6];
    bern5((x + 1.0f) * 0.5f, bx);
    bern5((y + 1.0f) * 0.5f, by);
    bern5((z + 1.0f) * 0.5f, bz);

    // Half-wave picks its 3 z-basis values (k<3 vs k>=3), duplicated to h2.
    h2 bzd[3];
#pragma unroll
    for (int kq = 0; kq < 3; ++kq) {
      const float bzf = half ? bz[kq + 3] : bz[kq];
      const _Float16 bzh = (_Float16)bzf;
      bzd[kq] = h2{bzh, bzh};
    }

    // Packed f16 outer product of x/y bases: bxy_pk[i*3+jh] = bx[i]*by[2jh..+1]
    h2 by_pk[3], bx_dup[6], bxy_pk[18];
#pragma unroll
    for (int jh = 0; jh < 3; ++jh)
      by_pk[jh] = h2{(_Float16)by[2 * jh], (_Float16)by[2 * jh + 1]};
#pragma unroll
    for (int i = 0; i < 6; ++i) {
      const _Float16 bh = (_Float16)bx[i];
      bx_dup[i] = h2{bh, bh};
    }
#pragma unroll
    for (int i = 0; i < 6; ++i)
#pragma unroll
      for (int jh = 0; jh < 3; ++jh)
        bxy_pk[i * 3 + jh] = bx_dup[i] * by_pk[jh];   // v_pk_mul_f16

    v8f acc = {};  // f32 accumulator (C = 0)
#pragma unroll
    for (int t = 0; t < 7; ++t) {
      // B chunk: lanes 0-15 supply K = 32t+0..15, lanes 16-31 K = 32t+16..31;
      // dword g holds slots f = 16t+2g, +1  ->  one v_pk_mul_f16 each.
      BReg b;
#pragma unroll
      for (int g = 0; g < 8; ++g) {
        const int f = 16 * t + 2 * g;
        if (f < 108)
          b.p[g] = bxy_pk[(f % 36) >> 1] * bzd[f / 36];
        else
          b.p[g] = h2{(_Float16)0.0f, (_Float16)0.0f};
      }
      acc = __builtin_amdgcn_wmma_f32_16x16x32_f16(
          /*neg_a=*/false, A[t], /*neg_b=*/false, b.v,
          /*c_mod=*/(short)0, acc, /*reuse_a=*/false, /*reuse_b=*/false);
    }

    // D layout: lanes 0-15, VGPR r holds M=r (= channel c), N = lane (= vertex).
    if (half == 0 && v < N) {
      out[3 * v + 0] = 2.0f * acc[0] - 1.0f;
      out[3 * v + 1] = 2.0f * acc[1] - 1.0f;
      out[3 * v + 2] = 2.0f * acc[2] - 1.0f;
    }
  }
}

extern "C" void kernel_launch(void* const* d_in, const int* in_sizes, int n_in,
                              void* d_out, int out_size, void* d_ws, size_t ws_size,
                              hipStream_t stream) {
  const float* verts = (const float*)d_in[0];  // [N,3] f32
  const float* ctrl  = (const float*)d_in[1];  // [3,6,6,6] f32
  float* out = (float*)d_out;                  // [1,N,3] f32
  const int N = in_sizes[0] / 3;
  const int numTiles = (N + 15) / 16;

  const int threads = 256;                 // 8 waves/block
  int blocks = (numTiles + 7) / 8;         // at least one tile per wave
  if (blocks > 1024) blocks = 1024;        // grid-stride amortizes setup
  if (blocks < 1) blocks = 1;
  ffd_bernstein_wmma<<<blocks, threads, 0, stream>>>(verts, ctrl, out, N, numTiles);
}
